// MultiHeadAttention_20246475833709
// MI455X (gfx1250) — matrix operand order repaired and checked by reading, not compiled, not run on hardware
//
#include <hip/hip_runtime.h>

// MHA forward for B=2, S=4096, D_MODEL=768, H=12, DK=64 on gfx1250 (wave32).
// f16 WMMA (v_wmma_f32_16x16x32_f16) with f32 accumulation; K/V tiles stream
// through LDS via GLOBAL_LOAD_ASYNC_TO_LDS_B128 with double buffering.
// V is stored pre-transposed ([B,H,DK,S]) so all WMMA fragment reads from LDS
// are contiguous b128 loads.

typedef __attribute__((ext_vector_type(16))) _Float16 v16h;
typedef __attribute__((ext_vector_type(8)))  float    v8f;

#define D_MODEL 768
#define N_HEADS 12
#define DKH     64
#define BATCH   2
#define SEQ     4096
#define MROWS   (BATCH * SEQ)        // 8192

// A-fragment (16x32 f16) K index for element j of the v16h, given lane half.
// lanes 0-15: K 0..15
// lanes 16-31: K 16..31
__device__ __forceinline__ int a_kidx(int j, int lh) {
  return (lh << 4) + j;
}

__device__ __forceinline__ v8f wmma16(v16h a, v16h b, v8f c) {
  return __builtin_amdgcn_wmma_f32_16x16x32_f16(false, a, false, b, (short)0, c,
                                                false, false);
}

// Async cache->LDS copy of 16 bytes per lane (ASYNCcnt-tracked, no VGPR data).
__device__ __forceinline__ void async_b128(unsigned lds_addr,
                                           const void* sbase,
                                           unsigned voff_bytes) {
  asm volatile("global_load_async_to_lds_b128 %0, %1, %2"
               :
               : "v"(lds_addr), "v"(voff_bytes), "s"(sbase)
               : "memory");
}

// ---------------------------------------------------------------------------
// Kernel 1: Q/K/V projections.  C = X @ W^T + b.  Q/K outputs go to f16
// head-split layout [B, H, S, DK]; V output is stored TRANSPOSED as
// [B, H, DK, S] so the attention kernel's P@V B-fragments are contiguous.
// Block = 256 threads (8 waves), tile 128(M) x 64(N), K-step 32.
// ---------------------------------------------------------------------------
__global__ __launch_bounds__(256) void qkv_proj_kernel(
    const float* __restrict__ Q, const float* __restrict__ K,
    const float* __restrict__ V, const float* __restrict__ Wq,
    const float* __restrict__ Wk, const float* __restrict__ Wv,
    const float* __restrict__ bq, const float* __restrict__ bk,
    const float* __restrict__ bv, _Float16* __restrict__ q16,
    _Float16* __restrict__ k16, _Float16* __restrict__ v16) {
  __shared__ __align__(16) _Float16 As[128 * 40];   // 128 rows x 32 k
  __shared__ __align__(16) _Float16 Bs[64 * 40];    // 64 n-rows x 32 k

  const float* X; const float* W; const float* bias; _Float16* out;
  if (blockIdx.z == 0)      { X = Q; W = Wq; bias = bq; out = q16; }
  else if (blockIdx.z == 1) { X = K; W = Wk; bias = bk; out = k16; }
  else                      { X = V; W = Wv; bias = bv; out = v16; }
  const bool v_transposed = (blockIdx.z == 2);

  const int t  = threadIdx.x;
  const int m0 = blockIdx.x * 128;
  const int n0 = blockIdx.y * 64;
  const int lane = t & 31, lh = lane >> 4, lr = lane & 15;
  const int wv = t >> 5;
  const int rm = (wv & 3) << 5;   // wave's 32-row strip
  const int cn = (wv >> 2) << 5;  // wave's 32-col strip

  v8f acc[2][2] = {};

  for (int k0 = 0; k0 < D_MODEL; k0 += 32) {
    // Stage A tile (f32 -> f16): 1024 float4 / 256 threads
#pragma unroll
    for (int i = 0; i < 4; ++i) {
      int idx = t + (i << 8);
      int r = idx >> 3, c = (idx & 7) << 2;
      const float* src = X + (size_t)(m0 + r) * D_MODEL + k0 + c;
      float4 f = *(const float4*)src;
      __builtin_prefetch(src + 32, 0, 3);  // speculative: OOB tail is dropped
      _Float16* d = &As[r * 40 + c];
      d[0] = (_Float16)f.x; d[1] = (_Float16)f.y;
      d[2] = (_Float16)f.z; d[3] = (_Float16)f.w;
    }
    // Stage B tile = W rows (f32 -> f16): 512 float4 / 256 threads
#pragma unroll
    for (int i = 0; i < 2; ++i) {
      int idx = t + (i << 8);
      int r = idx >> 3, c = (idx & 7) << 2;
      const float* src = W + (size_t)(n0 + r) * D_MODEL + k0 + c;
      float4 f = *(const float4*)src;
      __builtin_prefetch(src + 32, 0, 3);
      _Float16* d = &Bs[r * 40 + c];
      d[0] = (_Float16)f.x; d[1] = (_Float16)f.y;
      d[2] = (_Float16)f.z; d[3] = (_Float16)f.w;
    }
    __syncthreads();

    v16h a0, a1, b0, b1;
#pragma unroll
    for (int j = 0; j < 16; ++j) {
      int ak = a_kidx(j, lh);
      a0[j] = As[(rm + lr) * 40 + ak];
      a1[j] = As[(rm + 16 + lr) * 40 + ak];
      int bk = (lh << 4) + j;
      b0[j] = Bs[(cn + lr) * 40 + bk];
      b1[j] = Bs[(cn + 16 + lr) * 40 + bk];
    }
    acc[0][0] = wmma16(a0, b0, acc[0][0]);
    acc[0][1] = wmma16(a0, b1, acc[0][1]);
    acc[1][0] = wmma16(a1, b0, acc[1][0]);
    acc[1][1] = wmma16(a1, b1, acc[1][1]);
    __syncthreads();
  }

  const float bias0 = bias[n0 + cn + lr];
  const float bias1 = bias[n0 + cn + 16 + lr];
#pragma unroll
  for (int ai = 0; ai < 2; ++ai)
#pragma unroll
    for (int bi = 0; bi < 2; ++bi)
#pragma unroll
      for (int j = 0; j < 8; ++j) {
        int m = m0 + rm + (ai << 4) + j + (lh << 3);
        int n = n0 + cn + (bi << 4) + lr;
        float val = acc[ai][bi][j] + (bi ? bias1 : bias0);
        int bb = m >> 12, s = m & 4095, h = n >> 6, dk = n & 63;
        size_t o = v_transposed
                       ? (((size_t)bb * N_HEADS + h) * DKH + dk) * SEQ + s
                       : (((size_t)bb * N_HEADS + h) * SEQ + s) * DKH + dk;
        out[o] = (_Float16)val;
      }
}

// ---------------------------------------------------------------------------
// Kernel 2: flash attention.  Block = 128 threads (4 waves); each wave owns a
// 16-row query tile; block covers 64 query rows of one (b,h).  K ([kv][dk])
// and V^T ([dk][kv]) tiles are double-buffered in LDS and filled with
// global_load_async_to_lds_b128 so the next tile's HBM/L2 latency overlaps
// the current tile's 16 WMMAs.
// ---------------------------------------------------------------------------
__global__ __launch_bounds__(128) void flash_attn_kernel(
    const _Float16* __restrict__ q16, const _Float16* __restrict__ k16,
    const _Float16* __restrict__ v16, _Float16* __restrict__ o16) {
  __shared__ __align__(16) _Float16 Kbuf[2][64 * 64];  // [buf][kv][dk]
  __shared__ __align__(16) _Float16 Vbuf[2][64 * 64];  // [buf][dk][kv]
  __shared__ __align__(16) _Float16 Ps[4][16 * 64];    // per-wave P staging

  const int t = threadIdx.x;
  const int lane = t & 31, lh = lane >> 4, lr = lane & 15;
  const int wv = t >> 5;                      // wave 0..3
  const int bh = blockIdx.x;                  // 0..B*H-1
  const int q0 = blockIdx.y * 64 + wv * 16;   // query row base of this wave
  const size_t base = (size_t)bh * SEQ * DKH; // same offset for K and V^T
  const float scale = 0.125f;                 // 1/sqrt(64)

  // Issue one K tile (rows=kv) + one V^T tile (rows=dk) async (8 instr/wave).
  auto issue_tile = [&](int kv0, int buf) {
#pragma unroll
    for (int i = 0; i < 4; ++i) {
      int idx = t + (i << 7);
      int r = idx >> 3, c = (idx & 7) << 3;
      unsigned gk =
          (unsigned)((base + (size_t)(kv0 + r) * DKH + c) * sizeof(_Float16));
      unsigned gv =
          (unsigned)((base + (size_t)r * SEQ + kv0 + c) * sizeof(_Float16));
      async_b128((unsigned)(uintptr_t)&Kbuf[buf][r * 64 + c], k16, gk);
      async_b128((unsigned)(uintptr_t)&Vbuf[buf][r * 64 + c], v16, gv);
    }
  };

  // Load Q fragments once: A-layout, dk split 0..31 / 32..63
  v16h qA0, qA1;
#pragma unroll
  for (int j = 0; j < 16; ++j) {
    int d = a_kidx(j, lh);
    qA0[j] = q16[base + (size_t)(q0 + lr) * DKH + d];
    qA1[j] = q16[base + (size_t)(q0 + lr) * DKH + 32 + d];
  }

  v8f O[4] = {};
  float mrow[8], lrow[8];
#pragma unroll
  for (int j = 0; j < 8; ++j) { mrow[j] = -3.0e38f; lrow[j] = 0.0f; }

  issue_tile(0, 0);  // prefetch tile 0

  for (int kv0 = 0, it = 0; kv0 < SEQ; kv0 += 64, ++it) {
    const int cur = it & 1;
    if (kv0 + 64 < SEQ) {
      issue_tile(kv0 + 64, cur ^ 1);          // overlap: next tile in flight
      asm volatile("s_wait_asynccnt 0x8" ::: "memory");  // tile `cur` done
    } else {
      asm volatile("s_wait_asynccnt 0x0" ::: "memory");
    }
    __syncthreads();
    const _Float16* Ks = Kbuf[cur];  // [kv][dk]
    const _Float16* Vs = Vbuf[cur];  // [dk][kv]

    // scores S = Q @ K^T : four 16x16 column tiles, two K-steps (dk) each
    v8f sc[4];
#pragma unroll
    for (int ct = 0; ct < 4; ++ct) {
      v16h kb0, kb1;
      int n = ct * 16 + lr;  // kv row (column of S)
#pragma unroll
      for (int j = 0; j < 16; ++j) {
        int d = (lh << 4) + j;
        kb0[j] = Ks[n * 64 + d];
        kb1[j] = Ks[n * 64 + 32 + d];
      }
      v8f z = {};
      z = wmma16(qA0, kb0, z);
      z = wmma16(qA1, kb1, z);
      sc[ct] = z;
    }

    // Online softmax update (row r = j + 8*lh lives in this lane's half)
#pragma unroll
    for (int j = 0; j < 8; ++j) {
      float mx = -3.0e38f;
#pragma unroll
      for (int ct = 0; ct < 4; ++ct) mx = fmaxf(mx, sc[ct][j] * scale);
      mx = fmaxf(mx, __shfl_xor(mx, 8, 32));
      mx = fmaxf(mx, __shfl_xor(mx, 4, 32));
      mx = fmaxf(mx, __shfl_xor(mx, 2, 32));
      mx = fmaxf(mx, __shfl_xor(mx, 1, 32));
      float mnew  = fmaxf(mrow[j], mx);
      float alpha = __expf(mrow[j] - mnew);
      float ps = 0.0f;
#pragma unroll
      for (int ct = 0; ct < 4; ++ct) {
        float p = __expf(sc[ct][j] * scale - mnew);
        Ps[wv][(j + (lh << 3)) * 64 + ct * 16 + lr] = (_Float16)p;
        ps += p;
      }
      ps += __shfl_xor(ps, 8, 32);
      ps += __shfl_xor(ps, 4, 32);
      ps += __shfl_xor(ps, 2, 32);
      ps += __shfl_xor(ps, 1, 32);
      lrow[j] = lrow[j] * alpha + ps;
      mrow[j] = mnew;
#pragma unroll
      for (int dt = 0; dt < 4; ++dt) O[dt][j] *= alpha;
    }

    // O += P @ V : P re-laid as A fragments (kv split 0..31 / 32..63).
    // P is per-wave LDS; in-wave DS ordering (DScnt) covers the RAW.
    v16h pa0, pa1;
#pragma unroll
    for (int j = 0; j < 16; ++j) {
      int kk = a_kidx(j, lh);
      pa0[j] = Ps[wv][lr * 64 + kk];
      pa1[j] = Ps[wv][lr * 64 + 32 + kk];
    }
#pragma unroll
    for (int dt = 0; dt < 4; ++dt) {
      v16h vb0, vb1;
      int n = dt * 16 + lr;  // dk column; V^T row -> contiguous kv reads
#pragma unroll
      for (int j = 0; j < 16; ++j) {
        int kv = (lh << 4) + j;
        vb0[j] = Vs[n * 64 + kv];
        vb1[j] = Vs[n * 64 + 32 + kv];
      }
      O[dt] = wmma16(pa0, vb0, O[dt]);
      O[dt] = wmma16(pa1, vb1, O[dt]);
    }
    __syncthreads();  // all waves done with buf `cur` before it is re-filled
  }

  // Normalize and store merged-head f16 output [B, S, D_MODEL]
  const int b = bh / N_HEADS, h = bh % N_HEADS;
#pragma unroll
  for (int j = 0; j < 8; ++j) {
    float inv = 1.0f / lrow[j];
    int s = q0 + j + (lh << 3);
#pragma unroll
    for (int dt = 0; dt < 4; ++dt) {
      o16[((size_t)b * SEQ + s) * D_MODEL + h * DKH + dt * 16 + lr] =
          (_Float16)(O[dt][j] * inv);
    }
  }
}

// ---------------------------------------------------------------------------
// Kernel 3: output projection.  C = A16 @ Wo^T + bo, f32 result to d_out.
// A tile is a pure f16 copy -> async cache->LDS; W tile needs f32->f16 CVT
// and stages through VGPRs (overlapping the async transfer).
// ---------------------------------------------------------------------------
__global__ __launch_bounds__(256) void out_proj_kernel(
    const _Float16* __restrict__ A, const float* __restrict__ W,
    const float* __restrict__ bias, float* __restrict__ out) {
  __shared__ __align__(16) _Float16 As[128 * 40];
  __shared__ __align__(16) _Float16 Bs[64 * 40];

  const int t  = threadIdx.x;
  const int m0 = blockIdx.x * 128;
  const int n0 = blockIdx.y * 64;
  const int lane = t & 31, lh = lane >> 4, lr = lane & 15;
  const int wv = t >> 5;
  const int rm = (wv & 3) << 5;
  const int cn = (wv >> 2) << 5;

  v8f acc[2][2] = {};

  for (int k0 = 0; k0 < D_MODEL; k0 += 32) {
    // A tile: async f16 copy, 512 x b128 / 256 threads
#pragma unroll
    for (int i = 0; i < 2; ++i) {
      int idx = t + (i << 8);
      int r = idx >> 2, c = (idx & 3) << 3;
      unsigned go = (unsigned)(((size_t)(m0 + r) * D_MODEL + k0 + c) *
                               sizeof(_Float16));
      async_b128((unsigned)(uintptr_t)&As[r * 40 + c], A, go);
    }
    // B tile (f32 -> f16) through VGPRs while the async copy is in flight
#pragma unroll
    for (int i = 0; i < 2; ++i) {
      int idx = t + (i << 8);
      int r = idx >> 3, c = (idx & 7) << 2;
      const float* src = W + (size_t)(n0 + r) * D_MODEL + k0 + c;
      float4 f = *(const float4*)src;
      __builtin_prefetch(src + 32, 0, 3);
      _Float16* d = &Bs[r * 40 + c];
      d[0] = (_Float16)f.x; d[1] = (_Float16)f.y;
      d[2] = (_Float16)f.z; d[3] = (_Float16)f.w;
    }
    asm volatile("s_wait_asynccnt 0x0" ::: "memory");
    __syncthreads();

    v16h a0, a1, b0, b1;
#pragma unroll
    for (int j = 0; j < 16; ++j) {
      int ak = a_kidx(j, lh);
      a0[j] = As[(rm + lr) * 40 + ak];
      a1[j] = As[(rm + 16 + lr) * 40 + ak];
      int bk = (lh << 4) + j;
      b0[j] = Bs[(cn + lr) * 40 + bk];
      b1[j] = Bs[(cn + 16 + lr) * 40 + bk];
    }
    acc[0][0] = wmma16(a0, b0, acc[0][0]);
    acc[0][1] = wmma16(a0, b1, acc[0][1]);
    acc[1][0] = wmma16(a1, b0, acc[1][0]);
    acc[1][1] = wmma16(a1, b1, acc[1][1]);
    __syncthreads();
  }

  const float bias0 = bias[n0 + cn + lr];
  const float bias1 = bias[n0 + cn + 16 + lr];
#pragma unroll
  for (int ai = 0; ai < 2; ++ai)
#pragma unroll
    for (int bi = 0; bi < 2; ++bi)
#pragma unroll
      for (int j = 0; j < 8; ++j) {
        int m = m0 + rm + (ai << 4) + j + (lh << 3);
        int n = n0 + cn + (bi << 4) + lr;
        out[(size_t)m * D_MODEL + n] = acc[ai][bi][j] + (bi ? bias1 : bias0);
      }
}

// ---------------------------------------------------------------------------
extern "C" void kernel_launch(void* const* d_in, const int* in_sizes, int n_in,
                              void* d_out, int out_size, void* d_ws,
                              size_t ws_size, hipStream_t stream) {
  (void)in_sizes; (void)n_in; (void)out_size; (void)ws_size;

  const float* Q  = (const float*)d_in[0];
  const float* K  = (const float*)d_in[1];
  const float* V  = (const float*)d_in[2];
  const float* Wq = (const float*)d_in[3];
  const float* bq = (const float*)d_in[4];
  const float* Wk = (const float*)d_in[5];
  const float* bk = (const float*)d_in[6];
  const float* Wv = (const float*)d_in[7];
  const float* bv = (const float*)d_in[8];
  const float* Wo = (const float*)d_in[9];
  const float* bo = (const float*)d_in[10];
  float* out = (float*)d_out;

  // Workspace slab: 4 x (8192*768) f16 buffers = ~48 MB
  const size_t elems = (size_t)MROWS * D_MODEL;  // 6,291,456
  _Float16* q16 = (_Float16*)d_ws;
  _Float16* k16 = q16 + elems;
  _Float16* v16 = k16 + elems;   // stored transposed [B,H,DK,S]
  _Float16* a16 = v16 + elems;

  // 1) QKV projections (z selects q/k/v)
  qkv_proj_kernel<<<dim3(MROWS / 128, D_MODEL / 64, 3), 256, 0, stream>>>(
      Q, K, V, Wq, Wk, Wv, bq, bk, bv, q16, k16, v16);

  // 2) Flash attention: (B*H) x (S/64) blocks of 128 threads
  flash_attn_kernel<<<dim3(BATCH * N_HEADS, SEQ / 64), 128, 0, stream>>>(
      q16, k16, v16, a16);

  // 3) Output projection
  out_proj_kernel<<<dim3(MROWS / 128, D_MODEL / 64), 256, 0, stream>>>(
      a16, Wo, bo, out);
}
